// FeaturePropagationModule_56916906606963
// MI455X (gfx1250) — matrix operand order repaired and checked by reading, not compiled, not run on hardware
//
#include <hip/hip_runtime.h>

// ---------------- problem constants ----------------
#define BSZ   16
#define NCP   1024
#define NFP   4096
#define CIN   512
#define CSK   256
#define DIN   768          // CIN + CSK
#define COUT  512
#define MTOT  (BSZ * NFP)  // 65536
#define BN_EPS 1e-5f

// ---------------- WMMA vector types ----------------
typedef __attribute__((ext_vector_type(16))) __bf16 v16bf;
typedef __attribute__((ext_vector_type(8)))  float  v8f;

struct BPair { uint4 a, b; };
static __device__ __forceinline__ v16bf make_v16bf(uint4 lo, uint4 hi) {
  BPair p{lo, hi};
  return __builtin_bit_cast(v16bf, p);
}

// Async copy of 16 bytes (per lane) global -> LDS, tracked by ASYNCcnt.
// vdst = wave-relative LDS byte offset (low 32 bits of generic pointer),
// vaddr = 64-bit global address (GV mode, saddr = off).
static __device__ __forceinline__ void async_copy_b128(const void* gaddr,
                                                       const __bf16* laddr) {
  const unsigned lds_off = (unsigned)(unsigned long long)laddr;
  asm volatile("global_load_async_to_lds_b128 %0, %1, off"
               :
               : "v"(lds_off), "v"((unsigned long long)gaddr)
               : "memory");
}

static __device__ __forceinline__ void wait_asynccnt_4() {
  asm volatile("s_wait_asynccnt 0x4" ::: "memory");
}
static __device__ __forceinline__ void wait_asynccnt_0() {
  asm volatile("s_wait_asynccnt 0x0" ::: "memory");
}

// =====================================================================
// Kernel 1: kNN (K=3) + inverse-d^2 interpolation + concat with skip.
// One block (256 thr) per 256 fine points of one batch.
// Coarse positions staged in LDS (12 KB); feature gather is wave-per-point
// with lanes parallel over channels (coalesced global reads).
// =====================================================================
__global__ void __launch_bounds__(256)
fp_knn_interp_kernel(const float* __restrict__ x,
                     const float* __restrict__ pos,
                     const float* __restrict__ pos_skip,
                     const float* __restrict__ x_skip,
                     __bf16* __restrict__ h0) {
  __shared__ float lx[NCP], ly[NCP], lz[NCP];
  __shared__ int   s_idx[256][3];
  __shared__ float s_w[256][4];

  const int b     = blockIdx.x >> 4;          // 16 blocks per batch
  const int fbase = (blockIdx.x & 15) * 256;  // fine offset within batch
  const int t     = threadIdx.x;

  const float* pc = pos + (size_t)b * NCP * 3;
  for (int i = t; i < NCP; i += 256) {
    lx[i] = pc[i * 3 + 0];
    ly[i] = pc[i * 3 + 1];
    lz[i] = pc[i * 3 + 2];
  }
  __syncthreads();

  // ---- top-3 nearest per thread ----
  const size_t f = (size_t)b * NFP + fbase + t;
  const float px = pos_skip[f * 3 + 0];
  const float py = pos_skip[f * 3 + 1];
  const float pz = pos_skip[f * 3 + 2];
  float d0 = 1e30f, d1 = 1e30f, d2 = 1e30f;
  int   i0 = 0, i1 = 0, i2 = 0;
  for (int c = 0; c < NCP; ++c) {
    const float dx = px - lx[c], dy = py - ly[c], dz = pz - lz[c];
    const float d = dx * dx + dy * dy + dz * dz;
    if (d < d0)      { d2 = d1; i2 = i1; d1 = d0; i1 = i0; d0 = d; i0 = c; }
    else if (d < d1) { d2 = d1; i2 = i1; d1 = d;  i1 = c; }
    else if (d < d2) { d2 = d;  i2 = c; }
  }
  const float w0 = 1.f / fmaxf(d0, 1e-16f);
  const float w1 = 1.f / fmaxf(d1, 1e-16f);
  const float w2 = 1.f / fmaxf(d2, 1e-16f);
  s_idx[t][0] = i0; s_idx[t][1] = i1; s_idx[t][2] = i2;
  s_w[t][0] = w0; s_w[t][1] = w1; s_w[t][2] = w2;
  s_w[t][3] = 1.f / (w0 + w1 + w2);
  __syncthreads();

  // ---- wave-per-point gather (coalesced over channels) ----
  const int wave = t >> 5, lane = t & 31;
  const float* xb = x + (size_t)b * NCP * CIN;
  for (int p = wave; p < 256; p += 8) {
    const int   j0 = s_idx[p][0], j1 = s_idx[p][1], j2 = s_idx[p][2];
    const float ww0 = s_w[p][0], ww1 = s_w[p][1], ww2 = s_w[p][2], inv = s_w[p][3];
    const size_t fr = (size_t)b * NFP + fbase + p;
    __bf16* out = h0 + fr * DIN;
    const float* r0 = xb + (size_t)j0 * CIN;
    const float* r1 = xb + (size_t)j1 * CIN;
    const float* r2 = xb + (size_t)j2 * CIN;
    for (int c = lane; c < CIN; c += 32)
      out[c] = (__bf16)((ww0 * r0[c] + ww1 * r1[c] + ww2 * r2[c]) * inv);
    const float* sk = x_skip + fr * CSK;
    for (int c = lane; c < CSK; c += 32)
      out[CIN + c] = (__bf16)sk[c];
  }
}

// =====================================================================
// f32 -> bf16 converter (for weights)
// =====================================================================
__global__ void fp_cvt_bf16_kernel(const float* __restrict__ in,
                                   __bf16* __restrict__ out, int n) {
  const int i = blockIdx.x * 256 + threadIdx.x;
  if (i < n) out[i] = (__bf16)in[i];
}

// =====================================================================
// WMMA GEMM: C[M,N] = A[M,K](bf16) * B[K,N](bf16) + bias[N]
// Block tile 128x128, 256 threads = 8 waves; each wave does 64x32 via
// 4x2 tiles of v_wmma_f32_16x16x32_bf16. DOUBLE-buffered LDS filled with
// global_load_async_to_lds_b128 (ASYNCcnt-tracked) so the next K-slice
// streams from HBM while WMMAs consume the current one. Padded LDS row
// strides (A: 40 bf16, B: 136 bf16) for bank spread + b128 alignment.
// =====================================================================
#define AST 40
#define BST 136

__global__ void __launch_bounds__(256)
fp_gemm_bf16_kernel(const __bf16* __restrict__ A,
                    const __bf16* __restrict__ Bw,
                    const float* __restrict__ bias,
                    float* __restrict__ C,
                    int M, int N, int K) {
  __shared__ __bf16 sA[2][128 * AST];
  __shared__ __bf16 sB[2][32 * BST];

  const int t = threadIdx.x;
  const int ntiles_n = N >> 7;
  const int mtile = blockIdx.x / ntiles_n;
  const int ntile = blockIdx.x % ntiles_n;
  const int wave = t >> 5, lane = t & 31;
  const int wm = wave >> 2, wn = wave & 3;   // 2 waves (M) x 4 waves (N)
  const int m16 = lane & 15, kg = lane >> 4;

  v8f acc[4][2];
#pragma unroll
  for (int mi = 0; mi < 4; ++mi)
#pragma unroll
    for (int ni = 0; ni < 2; ++ni)
      acc[mi][ni] = (v8f){0.f, 0.f, 0.f, 0.f, 0.f, 0.f, 0.f, 0.f};

  // per-thread staging coordinates (32 bytes per tile each)
  const int ar = t >> 1, ac = (t & 1) * 16;         // A: 128 rows x 32 cols
  const int br = t >> 3, bc = (t & 7) * 16;         // B:  32 rows x 128 cols
  const __bf16* gA = A + (size_t)(mtile * 128 + ar) * K + ac;
  const __bf16* gB = Bw + (size_t)br * N + ntile * 128 + bc;

  const int nk = K >> 5;

  // prologue: stage K-slice 0 into buffer 0
  {
    async_copy_b128(gA, &sA[0][ar * AST + ac]);
    async_copy_b128(gA + 8, &sA[0][ar * AST + ac + 8]);
    async_copy_b128(gB, &sB[0][br * BST + bc]);
    async_copy_b128(gB + 8, &sB[0][br * BST + bc + 8]);
  }

  for (int ks = 0; ks < nk; ++ks) {
    const int d = ks & 1;
    if (ks + 1 < nk) {
      // prefetch next K-slice into the other buffer
      const int k1 = (ks + 1) << 5;
      async_copy_b128(gA + k1, &sA[d ^ 1][ar * AST + ac]);
      async_copy_b128(gA + k1 + 8, &sA[d ^ 1][ar * AST + ac + 8]);
      async_copy_b128(gB + (size_t)k1 * N, &sB[d ^ 1][br * BST + bc]);
      async_copy_b128(gB + (size_t)k1 * N + 8, &sB[d ^ 1][br * BST + bc + 8]);
      wait_asynccnt_4();   // my 4 loads for buffer d have landed
    } else {
      wait_asynccnt_0();
    }
    __syncthreads();       // all waves' loads for buffer d have landed

    // B fragments: lane <-> N column (lane&15), K {kb..kb+7} + {16+kb..16+kb+7}
    v16bf bf[2];
#pragma unroll
    for (int ni = 0; ni < 2; ++ni) {
      const __bf16* pb = &sB[d][wn * 32 + ni * 16 + m16];
#pragma unroll
      for (int s = 0; s < 16; ++s)
        bf[ni][s] = pb[((s < 8) ? (kg * 8 + s) : (16 + kg * 8 + s - 8)) * BST];
    }
#pragma unroll
    for (int mi = 0; mi < 4; ++mi) {
      // A fragment: lanes 0-15 rows M, K {kb..kb+7} + {16+kb..16+kb+7}
      const __bf16* pa = &sA[d][(wm * 64 + mi * 16 + m16) * AST + kg * 8];
      const v16bf af = make_v16bf(*(const uint4*)pa, *(const uint4*)(pa + 16));
#pragma unroll
      for (int ni = 0; ni < 2; ++ni)
        acc[mi][ni] = __builtin_amdgcn_wmma_f32_16x16x32_bf16(
            false, af, false, bf[ni], (short)0, acc[mi][ni], false, false);
    }
    __syncthreads();       // reads of buffer d done before it is refilled
  }

  // epilogue: + bias, scatter per C/D VGPR layout
  // VGPR j: lanes 0-15 -> M=j, lanes 16-31 -> M=j+8; N = lane&15
  const int nn = ntile * 128 + wn * 32 + (lane & 15);
#pragma unroll
  for (int ni = 0; ni < 2; ++ni) {
    const int ncol = nn + ni * 16;
    const float bv = bias[ncol];
#pragma unroll
    for (int mi = 0; mi < 4; ++mi)
#pragma unroll
      for (int j = 0; j < 8; ++j) {
        const int mrow = mtile * 128 + wm * 64 + mi * 16 + j + 8 * (lane >> 4);
        C[(size_t)mrow * N + ncol] = acc[mi][ni][j] + bv;
      }
  }
}

// =====================================================================
// BN column stats: partial sums / sumsq per column via atomics
// =====================================================================
__global__ void fp_zero_kernel(float* __restrict__ p, int n) {
  const int i = blockIdx.x * 256 + threadIdx.x;
  if (i < n) p[i] = 0.f;
}

__global__ void __launch_bounds__(256)
fp_colstats_kernel(const float* __restrict__ C, float* __restrict__ sums,
                   float* __restrict__ sumsq, int M, int N) {
  const int rows = M / gridDim.x;
  const int r0 = blockIdx.x * rows;
  for (int c = threadIdx.x; c < N; c += 256) {
    float s = 0.f, q = 0.f;
    for (int r = r0; r < r0 + rows; ++r) {
      const float v = C[(size_t)r * N + c];
      s += v; q += v * v;
    }
    atomicAdd(&sums[c], s);
    atomicAdd(&sumsq[c], q);
  }
}

__global__ void fp_bn_finalize_kernel(const float* __restrict__ sums,
                                      const float* __restrict__ sumsq,
                                      const float* __restrict__ g,
                                      const float* __restrict__ be,
                                      float* __restrict__ scale,
                                      float* __restrict__ shift,
                                      float invM, int N) {
  const int c = blockIdx.x * blockDim.x + threadIdx.x;
  if (c < N) {
    const float mu  = sums[c] * invM;
    const float var = sumsq[c] * invM - mu * mu;
    const float s   = g[c] * rsqrtf(var + BN_EPS);
    scale[c] = s;
    shift[c] = be[c] - mu * s;
  }
}

// BN affine + ReLU, emitting bf16 (next GEMM A operand)
__global__ void fp_bn_relu_bf16_kernel(const float* __restrict__ C,
                                       const float* __restrict__ scale,
                                       const float* __restrict__ shift,
                                       __bf16* __restrict__ out, size_t total) {
  const size_t i = (size_t)blockIdx.x * 256 + threadIdx.x;
  if (i < total) {
    const int c = (int)(i & (COUT - 1));
    out[i] = (__bf16)fmaxf(C[i] * scale[c] + shift[c], 0.f);
  }
}

// BN affine + ReLU, emitting f32 (final output)
__global__ void fp_bn_relu_f32_kernel(const float* __restrict__ C,
                                      const float* __restrict__ scale,
                                      const float* __restrict__ shift,
                                      float* __restrict__ out, size_t total) {
  const size_t i = (size_t)blockIdx.x * 256 + threadIdx.x;
  if (i < total) {
    const int c = (int)(i & (COUT - 1));
    out[i] = fmaxf(C[i] * scale[c] + shift[c], 0.f);
  }
}

// tail: pos_skip passthrough + batch_skip (int64) as float
__global__ void fp_tail_kernel(const float* __restrict__ pos_skip,
                               const long long* __restrict__ batch_skip,
                               float* __restrict__ out) {
  const int i = blockIdx.x * 256 + threadIdx.x;
  if (i < MTOT * 3) out[i] = pos_skip[i];
  if (i < MTOT) out[MTOT * 3 + i] = (float)batch_skip[i];
}

// =====================================================================
// launch
// =====================================================================
extern "C" void kernel_launch(void* const* d_in, const int* in_sizes, int n_in,
                              void* d_out, int out_size, void* d_ws, size_t ws_size,
                              hipStream_t stream) {
  const float*     x          = (const float*)d_in[0];
  const float*     pos        = (const float*)d_in[1];
  /* batch (d_in[2]) unused: contiguous equal-sized batches */
  const float*     x_skip     = (const float*)d_in[3];
  const float*     pos_skip   = (const float*)d_in[4];
  const long long* batch_skip = (const long long*)d_in[5];
  const float*     W1 = (const float*)d_in[6];
  const float*     b1 = (const float*)d_in[7];
  const float*     g1 = (const float*)d_in[8];
  const float*     be1 = (const float*)d_in[9];
  const float*     W2 = (const float*)d_in[10];
  const float*     b2 = (const float*)d_in[11];
  const float*     g2 = (const float*)d_in[12];
  const float*     be2 = (const float*)d_in[13];
  float* out = (float*)d_out;

  // ---- workspace layout (A2 aliases h0: h0 is dead before A2 is written) ----
  char* ws = (char*)d_ws;
  size_t o = 0;
  __bf16* h0  = (__bf16*)(ws + o);                 // [MTOT, DIN] bf16
  __bf16* A2  = (__bf16*)(ws + o);                 // [MTOT, COUT] bf16 (alias)
  o += (size_t)MTOT * DIN * 2;                     // 100,663,296
  __bf16* W1b = (__bf16*)(ws + o); o += (size_t)DIN * COUT * 2;
  __bf16* W2b = (__bf16*)(ws + o); o += (size_t)COUT * COUT * 2;
  float*  C1  = (float*)(ws + o);  o += (size_t)MTOT * COUT * 4;  // reused for C2
  float*  sums  = (float*)(ws + o); o += COUT * 4;
  float*  sumsq = (float*)(ws + o); o += COUT * 4;
  float*  scale = (float*)(ws + o); o += COUT * 4;
  float*  shift = (float*)(ws + o); o += COUT * 4;

  const size_t totalH = (size_t)MTOT * COUT;       // 33,554,432
  const int elemBlocks = (int)((totalH + 255) / 256);
  const float invM = 1.0f / (float)MTOT;

  // 1) kNN + interpolate + concat -> h0 (bf16)
  fp_knn_interp_kernel<<<MTOT / 256, 256, 0, stream>>>(x, pos, pos_skip, x_skip, h0);

  // 2) weights to bf16
  fp_cvt_bf16_kernel<<<(DIN * COUT + 255) / 256, 256, 0, stream>>>(W1, W1b, DIN * COUT);
  fp_cvt_bf16_kernel<<<(COUT * COUT + 255) / 256, 256, 0, stream>>>(W2, W2b, COUT * COUT);

  // 3) GEMM1: C1 = h0 @ W1 + b1
  fp_gemm_bf16_kernel<<<(MTOT / 128) * (COUT / 128), 256, 0, stream>>>(
      h0, W1b, b1, C1, MTOT, COUT, DIN);

  // 4) BN1 stats + finalize
  fp_zero_kernel<<<(2 * COUT + 255) / 256, 256, 0, stream>>>(sums, 2 * COUT);
  fp_colstats_kernel<<<128, 256, 0, stream>>>(C1, sums, sumsq, MTOT, COUT);
  fp_bn_finalize_kernel<<<2, 256, 0, stream>>>(sums, sumsq, g1, be1, scale, shift, invM, COUT);

  // 5) BN1 + ReLU -> A2 (bf16)
  fp_bn_relu_bf16_kernel<<<elemBlocks, 256, 0, stream>>>(C1, scale, shift, A2, totalH);

  // 6) GEMM2: C2 (=C1 buffer) = A2 @ W2 + b2
  fp_gemm_bf16_kernel<<<(MTOT / 128) * (COUT / 128), 256, 0, stream>>>(
      A2, W2b, b2, C1, MTOT, COUT, COUT);

  // 7) BN2 stats + finalize
  fp_zero_kernel<<<(2 * COUT + 255) / 256, 256, 0, stream>>>(sums, 2 * COUT);
  fp_colstats_kernel<<<128, 256, 0, stream>>>(C1, sums, sumsq, MTOT, COUT);
  fp_bn_finalize_kernel<<<2, 256, 0, stream>>>(sums, sumsq, g2, be2, scale, shift, invM, COUT);

  // 8) BN2 + ReLU -> d_out[0 : MTOT*COUT]
  fp_bn_relu_f32_kernel<<<elemBlocks, 256, 0, stream>>>(C1, scale, shift, out, totalH);

  // 9) pos_skip + batch_skip tail
  fp_tail_kernel<<<(MTOT * 3 + 255) / 256, 256, 0, stream>>>(
      pos_skip, batch_skip, out + totalH);
}
